// LiteFocusMSA_46849503265363
// MI455X (gfx1250) — matrix operand order repaired and checked by reading, not compiled, not run on hardware
//
#include <hip/hip_runtime.h>
#include <hip/hip_bf16.h>

// ---------------- types ----------------
typedef __bf16 bf16_t;
typedef bf16_t v16bf __attribute__((ext_vector_type(16)));
typedef float  v8f   __attribute__((ext_vector_type(8)));

union Frag16 {
    v16bf v;
    uint4 u4[2];
    unsigned int u[8];
};

__device__ __forceinline__ unsigned short f2bf(float f) {
    unsigned u = __float_as_uint(f);
    unsigned r = u + 0x7FFFu + ((u >> 16) & 1u);
    return (unsigned short)(r >> 16);
}
__device__ __forceinline__ unsigned f2bf_pk(float lo, float hi) {
    return (unsigned)f2bf(lo) | ((unsigned)f2bf(hi) << 16);
}

// LDS byte offset of a generic pointer into __shared__ (flat LDS aperture keeps
// the offset in addr[31:0] per CDNA5 aperture address calculation).
__device__ __forceinline__ unsigned lds_off_of(const void* p) {
    return (unsigned)(unsigned long long)p;
}

// CDNA5 async global->LDS copy, 16B per lane, tracked with ASYNCcnt.
// GVS addressing: mem = SGPR64 base + VGPR32 offset.
__device__ __forceinline__ void async_g2l_b128(unsigned ldsOff, unsigned gOff,
                                               const void* gBase) {
    asm volatile("global_load_async_to_lds_b128 %0, %1, %2"
                 :: "v"(ldsOff), "v"(gOff),
                    "s"((unsigned long long)(size_t)gBase)
                 : "memory");
}
__device__ __forceinline__ void wait_async0() {
    asm volatile("s_wait_asynccnt 0x0" ::: "memory");
}

// ---------------- constants ----------------
#define NPIX   6400          // 80*80
#define HH     80
#define TD3    1536
#define PROJK  1024
#define EPSF   1e-6f

// =====================================================================
// Prep: softplus(scale_param), BN fold, bf16-pack weights
// =====================================================================
__global__ __launch_bounds__(256) void lfmsa_prep_kernel(
    const float* __restrict__ w_qkv, const float* __restrict__ w_pw,
    const float* __restrict__ w_proj, const float* __restrict__ scale_param,
    const float* __restrict__ gamma, const float* __restrict__ beta,
    const float* __restrict__ mean, const float* __restrict__ var,
    unsigned short* __restrict__ wqkv_bf, unsigned short* __restrict__ wpw_bf,
    unsigned short* __restrict__ wproj_bf,
    float* __restrict__ scale_soft, float* __restrict__ bnscale,
    float* __restrict__ bnbias)
{
    int i = blockIdx.x * 256 + threadIdx.x;
    if (i < 98304)            wqkv_bf[i]           = f2bf(w_qkv[i]);
    else if (i < 196608)      wpw_bf[i - 98304]    = f2bf(w_pw[i - 98304]);
    else if (i < 262144)      wproj_bf[i - 196608] = f2bf(w_proj[i - 196608]);
    if (i < 64) {
        float x = scale_param[i];
        scale_soft[i] = (x > 20.f) ? x : log1pf(expf(x));
        float inv = gamma[i] * rsqrtf(var[i] + 1e-5f);
        bnscale[i] = inv;
        bnbias[i]  = beta[i] - mean[i] * inv;
    }
}

// =====================================================================
// WMMA GEMM:  C[m][n] = sum_k A_bf16[m][k] * B[k][n]  (+ row scale/bias)
//   A: bf16 row-major [Mg][K] per group   -> async DMA to LDS
//   B: BNK==false: f32 [K][N] row stride ldN -> VALU convert to LDS
//      BNK==true : bf16 [N][K] row stride K  -> async DMA to LDS
//   Double-buffered LDS pipeline: tile k+1 DMA overlaps tile k WMMAs,
//   s_wait_asynccnt only at the consume point. blockIdx.z = batch*nG + g.
// =====================================================================
template<int BM, int BN, int WGM, int WGN, bool BNK>
__global__ __launch_bounds__(256) void wmma_gemm_kernel(
    const unsigned short* __restrict__ A, const void* __restrict__ Bsrc,
    float* __restrict__ C, int K, int ldN,
    long long aG, long long bG, long long cG,
    long long aB, long long bB, long long cB, int nGroups,
    const float* __restrict__ rowScale, const float* __restrict__ rowBias)
{
    constexpr int MT   = BM / (16 * WGM);
    constexpr int NT   = BN / (16 * WGN);
    constexpr int AS   = 40;                 // padded LDS row stride (halves)
    constexpr int BUFS = (BM + BN) * AS;     // halves per buffer
    __shared__ unsigned short lds[2 * BUFS];

    const int tid   = threadIdx.x;
    const int lane  = tid & 31;
    const int w     = tid >> 5;
    const int wm    = w / WGN, wn = w % WGN;
    const int laneL = lane & 15, laneH = lane >> 4;

    const int z = blockIdx.z;
    const int g = z % nGroups, bb = z / nGroups;
    const unsigned short* Ap = A + (size_t)g * aG + (size_t)bb * aB
                                 + (size_t)blockIdx.y * BM * K;
    const float*          BpF = (const float*)Bsrc
                                 + (size_t)g * bG + (size_t)bb * bB;
    const unsigned short* BpH = (const unsigned short*)Bsrc
                                 + (size_t)g * bG + (size_t)bb * bB;
    float* Cp = C + (size_t)g * cG + (size_t)bb * cB;
    const int n0     = blockIdx.x * BN;
    const int mBlock = blockIdx.y * BM;

    // hoisted per-thread staging coordinates
    const int arow0 = tid >> 2, acc4 = tid & 3;          // A: 16B chunk coords
    const int bnn   = tid % BN, bkp = tid / BN;          // B(f32): n, k-pair base
    const float* BtBase = BpF + (size_t)(2 * bkp) * ldN + n0 + bnn;

    v8f acc[MT][NT];
    const v8f vzero = {0.f, 0.f, 0.f, 0.f, 0.f, 0.f, 0.f, 0.f};
#pragma unroll
    for (int im = 0; im < MT; ++im)
#pragma unroll
        for (int in = 0; in < NT; ++in) acc[im][in] = vzero;

    const int nK = K >> 5;

    // ---- staging of K-tile kt into buffer buf ----
    auto stage = [&](int kt, int buf) {
        unsigned short* Al = lds + buf * BUFS;
        unsigned short* Bl = Al + BM * AS;
        const int k0 = kt << 5;
        // A tile: BM rows x 64B, async DMA in 16B chunks
#pragma unroll
        for (int j = 0; j < (BM * 4) / 256; ++j) {
            int r = arow0 + j * 64;
            unsigned gOff = (unsigned)((size_t)r * K + k0) * 2u + acc4 * 16u;
            async_g2l_b128(lds_off_of(Al + r * AS + acc4 * 8), gOff, Ap);
        }
        if constexpr (BNK) {
            // B tile: bf16 [n][k] rows, async DMA in 16B chunks
#pragma unroll
            for (int j = 0; j < (BN * 4) / 256; ++j) {
                int r = arow0 + j * 64;
                unsigned gOff = (unsigned)((size_t)(n0 + r) * K + k0) * 2u
                                + acc4 * 16u;
                async_g2l_b128(lds_off_of(Bl + r * AS + acc4 * 8), gOff, BpH);
            }
        } else {
            // B tile: f32 [k][n] -> bf16 [n][k]; one packed b32 store per pair
            const float* bt = BtBase + (size_t)k0 * ldN;
            unsigned int* blu = (unsigned int*)(Bl + bnn * AS + 2 * bkp);
#pragma unroll
            for (int j = 0; j < (16 * BN) / 256; ++j) {
                float f0 = bt[(size_t)(4 * j) * ldN];
                float f1 = bt[(size_t)(4 * j + 1) * ldN];
                blu[2 * j] = f2bf_pk(f0, f1);
            }
            if (kt + 1 < nK)   // prefetch next K-slab (global_prefetch_b8)
                __builtin_prefetch(BtBase + (size_t)(k0 + 32) * ldN, 0, 1);
        }
    };

    // ---- prologue ----
    stage(0, 0);
    wait_async0();
    __syncthreads();

    // ---- main pipeline ----
    for (int kt = 0; kt < nK; ++kt) {
        const int cur = kt & 1;
        if (kt + 1 < nK) stage(kt + 1, cur ^ 1);   // DMA overlaps WMMAs below

        unsigned short* Al = lds + cur * BUFS;
        unsigned short* Bl = Al + BM * AS;

        Frag16 afr[MT], bfr[NT];
#pragma unroll
        for (int im = 0; im < MT; ++im) {
            int row = wm * MT * 16 + im * 16 + laneL;
            afr[im].u4[0] = *(const uint4*)(Al + row * AS + 8 * laneH);
            afr[im].u4[1] = *(const uint4*)(Al + row * AS + 16 + 8 * laneH);
        }
#pragma unroll
        for (int in = 0; in < NT; ++in) {
            int rn = wn * NT * 16 + in * 16 + laneL;
            bfr[in].u4[0] = *(const uint4*)(Bl + rn * AS + 8 * laneH);
            bfr[in].u4[1] = *(const uint4*)(Bl + rn * AS + 16 + 8 * laneH);
        }
#pragma unroll
        for (int im = 0; im < MT; ++im)
#pragma unroll
            for (int in = 0; in < NT; ++in)
                acc[im][in] = __builtin_amdgcn_wmma_f32_16x16x32_bf16(
                    false, afr[im].v, false, bfr[in].v,
                    (short)0, acc[im][in], false, false);

        if (kt + 1 < nK) {
            wait_async0();     // consume point for tile kt+1
            __syncthreads();
        }
    }

    // ---- epilogue: C layout VGPR r -> M = r + 8*laneH, N = laneL ----
#pragma unroll
    for (int im = 0; im < MT; ++im)
#pragma unroll
        for (int in = 0; in < NT; ++in) {
            int mBase = mBlock + wm * MT * 16 + im * 16 + 8 * laneH;
            int n     = n0 + wn * NT * 16 + in * 16 + laneL;
#pragma unroll
            for (int r = 0; r < 8; ++r) {
                int   mm = mBase + r;
                float vv = acc[im][in][r];
                if (rowScale) vv = vv * rowScale[mm] + rowBias[mm];
                Cp[(size_t)mm * ldN + n] = vv;
            }
        }
}

// =====================================================================
// Depthwise 5x5, pad 2, groups = channels
// =====================================================================
__global__ __launch_bounds__(256) void lfmsa_dwconv5_kernel(
    const float* __restrict__ in, const float* __restrict__ wdw,
    float* __restrict__ out)
{
    int n  = blockIdx.x * 256 + threadIdx.x;
    int ch = blockIdx.y;
    int b  = blockIdx.z;
    int py = n / HH, px = n % HH;
    const float* ip = in  + ((size_t)b * TD3 + ch) * NPIX;
    const float* wp = wdw + ch * 25;
    float acc = 0.f;
#pragma unroll
    for (int dy = -2; dy <= 2; ++dy) {
        int yy = py + dy;
        if (yy < 0 || yy >= HH) continue;
#pragma unroll
        for (int dx = -2; dx <= 2; ++dx) {
            int xx = px + dx;
            if (xx < 0 || xx >= HH) continue;
            acc += wp[(dy + 2) * 5 + (dx + 2)] * ip[yy * HH + xx];
        }
    }
    out[((size_t)b * TD3 + ch) * NPIX + n] = acc;
}

// =====================================================================
// Attention reduce: per (b,s) compute ksum[64] and kv[8 heads][8][8]
// =====================================================================
__global__ __launch_bounds__(256) void lfmsa_attn_reduce_kernel(
    const float* __restrict__ qkv, const float* __restrict__ agg,
    const float* __restrict__ scale_soft,
    float* __restrict__ ksum_ws, float* __restrict__ kv_ws)
{
    const int bs = blockIdx.x;              // 0..31
    const int b  = bs >> 4, s = bs & 15;
    const float* src = (s < 8) ? qkv : agg;
    const int sp = s & 7;
    const size_t baseK = ((size_t)b * TD3 + sp * 192 + 64)  * NPIX;
    const size_t baseV = ((size_t)b * TD3 + sp * 192 + 128) * NPIX;

    __shared__ float s_scale[64];
    __shared__ float kf[32][66];
    __shared__ float vv[32][66];
    __shared__ float fac[32];

    int tid = threadIdx.x;
    if (tid < 64) s_scale[tid] = scale_soft[tid];

    float kvacc0 = 0.f, kvacc1 = 0.f, ksumacc = 0.f;
    const int e0 = tid * 2;
    const int h  = e0 >> 6, cc = (e0 >> 3) & 7, dd0 = e0 & 7;

    for (int n0 = 0; n0 < NPIX; n0 += 32) {
        __syncthreads();
#pragma unroll
        for (int j = 0; j < 8; ++j) {
            int i = tid + j * 256;
            int d = i >> 5, ni = i & 31;
            float kl = src[baseK + (size_t)d * NPIX + n0 + ni];
            kf[ni][d] = (fmaxf(kl, 0.f) + EPSF) / s_scale[d];
            vv[ni][d] = src[baseV + (size_t)d * NPIX + n0 + ni];
        }
        __syncthreads();
        if (tid < 32) {               // focus factor: ||t|| / ||t^3||
            float s1 = 0.f, s3 = 0.f;
            for (int d = 0; d < 64; ++d) {
                float t = kf[tid][d], t2 = t * t, t3 = t2 * t;
                s1 += t2; s3 += t3 * t3;
            }
            fac[tid] = sqrtf(s1) * rsqrtf(fmaxf(s3, 1e-30f));
        }
        __syncthreads();
#pragma unroll
        for (int j = 0; j < 8; ++j) {
            int i = tid + j * 256;
            int d = i >> 5, ni = i & 31;
            float t = kf[ni][d];
            kf[ni][d] = t * t * t * fac[ni];
        }
        __syncthreads();
        for (int ni = 0; ni < 32; ++ni) {
            float kfv = kf[ni][h * 8 + cc];
            kvacc0 += kfv * vv[ni][h * 8 + dd0];
            kvacc1 += kfv * vv[ni][h * 8 + dd0 + 1];
        }
        if (tid < 64)
            for (int ni = 0; ni < 32; ++ni) ksumacc += kf[ni][tid];
    }
    kv_ws[bs * 512 + e0]     = kvacc0;
    kv_ws[bs * 512 + e0 + 1] = kvacc1;
    if (tid < 64) ksum_ws[bs * 64 + tid] = ksumacc;
}

// =====================================================================
// Attention apply: att = (q_foc . kv) / (q_foc . ksum + eps) + LePE
// One wave per pixel, 2 dims per lane (d and d+32).
// Output written as bf16 in [b][n][1024] (proj GEMM B layout, coalesced).
// =====================================================================
__global__ __launch_bounds__(256) void lfmsa_attn_apply_kernel(
    const float* __restrict__ qkv, const float* __restrict__ agg,
    const float* __restrict__ scale_soft,
    const float* __restrict__ ksum_ws, const float* __restrict__ kv_ws,
    const float* __restrict__ w_dwc, const float* __restrict__ b_dwc,
    unsigned short* __restrict__ attout)
{
    const int bs = blockIdx.y;
    const int b  = bs >> 4, s = bs & 15;
    const float* src = (s < 8) ? qkv : agg;
    const int sp = s & 7;
    const size_t baseQ   = ((size_t)b * TD3 + sp * 192) * NPIX;
    const size_t baseV   = baseQ + (size_t)128 * NPIX;
    const size_t outBase = (size_t)b * NPIX * PROJK + (size_t)s * 64;

    __shared__ float s_scale[64], s_ksum[64], s_kv[512], s_w[200], s_b[8];
    int tid = threadIdx.x;
    if (tid < 64) { s_scale[tid] = scale_soft[tid]; s_ksum[tid] = ksum_ws[bs * 64 + tid]; }
    s_kv[tid]       = kv_ws[bs * 512 + tid];
    s_kv[tid + 256] = kv_ws[bs * 512 + tid + 256];
    if (tid < 200) s_w[tid] = w_dwc[tid];
    if (tid < 8)   s_b[tid] = b_dwc[tid];
    __syncthreads();

    const int lane = tid & 31, w = tid >> 5;
    const int d0 = lane, d1 = lane + 32;
    const int h0 = lane >> 3, h1 = h0 + 4;
    const int dd = lane & 7;
    const int baseLane = lane & 24;

    for (int it = 0; it < 8; ++it) {
        int n = blockIdx.x * 64 + w * 8 + it;
        float q0 = src[baseQ + (size_t)d0 * NPIX + n];
        float q1 = src[baseQ + (size_t)d1 * NPIX + n];
        q0 = (fmaxf(q0, 0.f) + EPSF) / s_scale[d0];
        q1 = (fmaxf(q1, 0.f) + EPSF) / s_scale[d1];
        float q03 = q0 * q0 * q0, q13 = q1 * q1 * q1;
        float p2 = q0 * q0 + q1 * q1;
        float p6 = q03 * q03 + q13 * q13;
#pragma unroll
        for (int m = 16; m >= 1; m >>= 1) {
            p2 += __shfl_xor(p2, m, 32);
            p6 += __shfl_xor(p6, m, 32);
        }
        float facn = sqrtf(p2) * rsqrtf(fmaxf(p6, 1e-30f));
        float qf0 = q03 * facn, qf1 = q13 * facn;

        float z0 = 0.f, z1 = 0.f, a0 = 0.f, a1 = 0.f;
#pragma unroll
        for (int c = 0; c < 8; ++c) {
            float g0 = __shfl(qf0, baseLane + c, 32);
            float g1 = __shfl(qf1, baseLane + c, 32);
            z0 += g0 * s_ksum[h0 * 8 + c];
            z1 += g1 * s_ksum[h1 * 8 + c];
            a0 += g0 * s_kv[h0 * 64 + c * 8 + dd];
            a1 += g1 * s_kv[h1 * 64 + c * 8 + dd];
        }
        a0 /= (z0 + EPSF);
        a1 /= (z1 + EPSF);

        // LePE depthwise 5x5 on v, filter/bias index = dd
        int py = n / HH, px = n % HH;
        float l0 = s_b[dd], l1 = s_b[dd];
#pragma unroll
        for (int dy = -2; dy <= 2; ++dy) {
            int yy = py + dy;
            if (yy < 0 || yy >= HH) continue;
#pragma unroll
            for (int dx = -2; dx <= 2; ++dx) {
                int xx = px + dx;
                if (xx < 0 || xx >= HH) continue;
                float wv = s_w[dd * 25 + (dy + 2) * 5 + (dx + 2)];
                int nn2 = yy * HH + xx;
                l0 += wv * src[baseV + (size_t)d0 * NPIX + nn2];
                l1 += wv * src[baseV + (size_t)d1 * NPIX + nn2];
            }
        }
        attout[outBase + (size_t)n * PROJK + d0] = f2bf(a0 + l0);
        attout[outBase + (size_t)n * PROJK + d1] = f2bf(a1 + l1);
    }
}

// =====================================================================
// Launch
// =====================================================================
extern "C" void kernel_launch(void* const* d_in, const int* in_sizes, int n_in,
                              void* d_out, int out_size, void* d_ws, size_t ws_size,
                              hipStream_t stream) {
    const float* x        = (const float*)d_in[0];
    const float* w_qkv    = (const float*)d_in[1];
    const float* w_aggdw  = (const float*)d_in[2];
    const float* w_aggpw  = (const float*)d_in[3];
    const float* scale_p  = (const float*)d_in[4];
    const float* w_dwc    = (const float*)d_in[5];
    const float* b_dwc    = (const float*)d_in[6];
    const float* w_proj   = (const float*)d_in[7];
    const float* bn_gamma = (const float*)d_in[8];
    const float* bn_beta  = (const float*)d_in[9];
    const float* bn_mean  = (const float*)d_in[10];
    const float* bn_var   = (const float*)d_in[11];
    float* out = (float*)d_out;

    float* ws = (float*)d_ws;
    float* qkv    = ws;                       // 2*1536*6400 f32
    float* aggdw  = ws + 19660800;            // 2*1536*6400 f32 (later aliased attout)
    float* aggpw  = ws + 39321600;            // 2*1536*6400 f32
    float* ksum   = ws + 58982400;            // 32*64
    float* kvbuf  = ws + 58984448;            // 32*512
    float* sscale = ws + 59000832;            // 64
    float* bnsc   = ws + 59000896;            // 64
    float* bnbi   = ws + 59000960;            // 64
    unsigned short* wqkv_bf  = (unsigned short*)(ws + 59001024);
    unsigned short* wpw_bf   = wqkv_bf + 98304;
    unsigned short* wproj_bf = wpw_bf + 98304;
    // attout: bf16 [b][n][1024], aliased over aggdw (dead after pw GEMM)
    unsigned short* attout_bf = (unsigned short*)aggdw;

    // 0) prep
    lfmsa_prep_kernel<<<1024, 256, 0, stream>>>(
        w_qkv, w_aggpw, w_proj, scale_p, bn_gamma, bn_beta, bn_mean, bn_var,
        wqkv_bf, wpw_bf, wproj_bf, sscale, bnsc, bnbi);

    // 1) qkv 1x1 conv: M=1536, K=64, N=6400, batch 2
    wmma_gemm_kernel<128, 128, 4, 2, false><<<dim3(50, 12, 2), 256, 0, stream>>>(
        wqkv_bf, x, qkv, 64, NPIX,
        0LL, 0LL, 0LL,
        0LL, (long long)64 * NPIX, (long long)TD3 * NPIX, 1, nullptr, nullptr);

    // 2) depthwise 5x5 on qkv
    lfmsa_dwconv5_kernel<<<dim3(25, TD3, 2), 256, 0, stream>>>(qkv, w_aggdw, aggdw);

    // 3) grouped 1x1 (24 groups of 64x64), batch 2
    wmma_gemm_kernel<64, 128, 2, 4, false><<<dim3(50, 1, 48), 256, 0, stream>>>(
        wpw_bf, aggdw, aggpw, 64, NPIX,
        (long long)64 * 64, (long long)64 * NPIX, (long long)64 * NPIX,
        0LL, (long long)TD3 * NPIX, (long long)TD3 * NPIX, 24, nullptr, nullptr);

    // 4) attention reductions: ksum + kv per (b,s)
    lfmsa_attn_reduce_kernel<<<32, 256, 0, stream>>>(qkv, aggpw, sscale, ksum, kvbuf);

    // 5) attention apply + LePE -> attout bf16 [b][n][1024]
    lfmsa_attn_apply_kernel<<<dim3(100, 32), 256, 0, stream>>>(
        qkv, aggpw, sscale, ksum, kvbuf, w_dwc, b_dwc, attout_bf);

    // 6) proj 1x1 conv + BN: M=64, K=1024, N=6400, batch 2 (B = bf16 [n][k])
    wmma_gemm_kernel<64, 128, 2, 4, true><<<dim3(50, 1, 2), 256, 0, stream>>>(
        wproj_bf, attout_bf, out, PROJK, NPIX,
        0LL, 0LL, 0LL,
        0LL, (long long)NPIX * PROJK, (long long)64 * NPIX, 1, bnsc, bnbi);
}
